// MelPerformer_87952340287576
// MI455X (gfx1250) — matrix operand order repaired and checked by reading, not compiled, not run on hardware
//
#include <hip/hip_runtime.h>
#include <hip/hip_bf16.h>
#include <math.h>

typedef __attribute__((ext_vector_type(16))) _Float16 v16h;
typedef __attribute__((ext_vector_type(8)))  _Float16 v8h;
typedef __attribute__((ext_vector_type(4)))  _Float16 v4h;
typedef __attribute__((ext_vector_type(8)))  float    v8f;
typedef __attribute__((ext_vector_type(4)))  float    v4f;

#define N_BATCH  8
#define SEQ      3000
#define N_MELS   64
#define DIM      256
#define HEADS    4
#define DHEAD    64
#define MFEAT    512
#define N_LAYERS 4
#define FF_HID   1024
#define ROWS_TOT (N_BATCH*SEQ)                 /* 24000 */
#define DN_SCALE 0.35355339059327373f          /* 64^-0.25 */
#define RATIO    0.044194173824159216f         /* 512^-0.5 */
#define EPS_KF   1e-4f

static inline int cdiv(long a, long b) { return (int)((a + b - 1) / b); }

// ---------------------------------------------------------------------------
// CDNA5 async global->LDS copy (ASYNCcnt-tracked). vdst = LDS byte address
// (low 32 bits of a generic shared pointer), vaddr = 64-bit global address.
// ---------------------------------------------------------------------------
__device__ __forceinline__ void async_b128(void* lds_ptr, const void* gptr) {
    unsigned loff = (unsigned)(unsigned long long)lds_ptr;
    unsigned long long ga = (unsigned long long)gptr;
    asm volatile("global_load_async_to_lds_b128 %0, %1, off"
                 :: "v"(loff), "v"(ga) : "memory");
}
__device__ __forceinline__ void wait_async0() {
    asm volatile("s_wait_asynccnt 0x0" ::: "memory");
}

// ---------------------------------------------------------------------------
// Generic WMMA GEMM: C = op(A) * op(B) (+bias) (+resid) (*rowscale)
//   A: Mdim x Kdim (AT => stored Kdim x Mdim), f32 in memory
//   B: Kdim x Ndim (BT => stored Ndim x Kdim), f32 in memory
//   f32 -> f16 staged through LDS, v_wmma_f32_16x16x32_f16, f32 accumulate.
// Tile: BM=128 x BN=64 x BK=32, 128 threads (4 waves), 8 WMMA / wave / K-step.
// Full interior tiles of non-transposed A are fetched with
// global_load_async_to_lds_b128 double-buffered against WMMA compute.
// OUTMODE 0: C[z*sC + m*Ndim + n]
// OUTMODE 1: split heads: b=m/Nseq, n0=m%Nseq, h=n/64 -> C[((b*Hm+h)*Nseq+n0)*64+d]
// OUTMODE 2: merge heads (z = head): C[m*(Hm*Ndim) + z*Ndim + n]
// ---------------------------------------------------------------------------
template<bool AT, bool BT, int OUTMODE>
__global__ __launch_bounds__(128) void gemm_wmma(
    const float* __restrict__ A, const float* __restrict__ Bm,
    const float* __restrict__ bias, const float* __restrict__ resid,
    const float* __restrict__ rowscale, float* __restrict__ C,
    int Mdim, int Ndim, int Kdim,
    long sA, long sB, long sC, int Hm, int Nseq, float ascale)
{
    constexpr int BM = 128, BN = 64, BK = 32, LDA = BK + 8, LDB = BK + 8;
    __shared__ __align__(16) _Float16 As[BM * LDA];
    __shared__ __align__(16) _Float16 Bs[BN * LDB];
    __shared__ __align__(16) float    rawA[BM * BK];

    const int tid  = threadIdx.x;
    const int wave = tid >> 5;
    const int lane = tid & 31;
    const int z    = blockIdx.z;
    const float* Ab = A + (long)z * sA;
    const float* Bb = Bm + (long)z * sB;
    const int m0 = blockIdx.y * BM;
    const int n0 = blockIdx.x * BN;

    v8f acc[2][4] = {};

    // Async pipeline eligible: contiguous A rows, all K tiles full, full M tile.
    const bool pipeA = (!AT) && ((Kdim & (BK - 1)) == 0) && (m0 + BM <= Mdim);

    if (pipeA) {
        #pragma unroll
        for (int it = 0; it < 8; ++it) {                 // 128 thr * 8 * 4 floats = tile
            int id = it * 128 + tid;
            int i = id >> 3, j4 = id & 7;
            async_b128(&rawA[id * 4], Ab + (long)(m0 + i) * Kdim + j4 * 4);
        }
    }

    for (int k0 = 0; k0 < Kdim; k0 += BK) {
        if (pipeA) wait_async0();
        __syncthreads();   // async data visible to all waves; prev compute done

        // ---------------- stage A -> As (f16) ----------------
        if (pipeA) {
            #pragma unroll
            for (int it = 0; it < 8; ++it) {
                int id = it * 128 + tid;
                int i = id >> 3, j4 = id & 7;
                v4f vv = *(const v4f*)&rawA[id * 4];
                v4h hh;
                hh[0] = (_Float16)(vv[0] * ascale); hh[1] = (_Float16)(vv[1] * ascale);
                hh[2] = (_Float16)(vv[2] * ascale); hh[3] = (_Float16)(vv[3] * ascale);
                *(v4h*)&As[i * LDA + j4 * 4] = hh;
            }
        } else if (!AT) {
            if (m0 + BM <= Mdim && k0 + BK <= Kdim) {
                #pragma unroll
                for (int it = 0; it < 8; ++it) {
                    int id = it * 128 + tid;
                    int i = id >> 3, j4 = id & 7;
                    const float* gp = Ab + (long)(m0 + i) * Kdim + k0 + j4 * 4;
                    if (it == 0 && k0 + BK < Kdim)
                        __builtin_prefetch((const void*)(gp + BK), 0, 0);
                    v4f vv = *(const v4f*)gp;
                    v4h hh;
                    hh[0] = (_Float16)(vv[0] * ascale); hh[1] = (_Float16)(vv[1] * ascale);
                    hh[2] = (_Float16)(vv[2] * ascale); hh[3] = (_Float16)(vv[3] * ascale);
                    *(v4h*)&As[i * LDA + j4 * 4] = hh;
                }
            } else {
                #pragma unroll
                for (int it = 0; it < 32; ++it) {        // branchless clamped edge path
                    int id = it * 128 + tid;
                    int i = id >> 5, j = id & 31;
                    int gm = m0 + i, gk = k0 + j;
                    bool ok = (gm < Mdim) && (gk < Kdim);
                    int cm = gm < Mdim ? gm : Mdim - 1;
                    int ck = gk < Kdim ? gk : Kdim - 1;
                    float vA = Ab[(long)cm * Kdim + ck];
                    As[i * LDA + j] = (_Float16)(ok ? vA * ascale : 0.f);
                }
            }
        } else { // AT: A stored Kdim x Mdim; vectorize along m
            if (m0 + BM <= Mdim && k0 + BK <= Kdim) {
                #pragma unroll
                for (int it = 0; it < 8; ++it) {
                    int id = it * 128 + tid;
                    int j = id >> 5, i4 = id & 31;
                    const float* gp = Ab + (long)(k0 + j) * Mdim + m0 + i4 * 4;
                    if (it == 0 && k0 + BK < Kdim)
                        __builtin_prefetch((const void*)(gp + (long)BK * Mdim), 0, 0);
                    v4f vv = *(const v4f*)gp;
                    As[(i4 * 4 + 0) * LDA + j] = (_Float16)(vv[0] * ascale);
                    As[(i4 * 4 + 1) * LDA + j] = (_Float16)(vv[1] * ascale);
                    As[(i4 * 4 + 2) * LDA + j] = (_Float16)(vv[2] * ascale);
                    As[(i4 * 4 + 3) * LDA + j] = (_Float16)(vv[3] * ascale);
                }
            } else {
                #pragma unroll
                for (int it = 0; it < 32; ++it) {
                    int id = it * 128 + tid;
                    int j = id >> 7, i = id & 127;
                    int gm = m0 + i, gk = k0 + j;
                    bool ok = (gm < Mdim) && (gk < Kdim);
                    int cm = gm < Mdim ? gm : Mdim - 1;
                    int ck = gk < Kdim ? gk : Kdim - 1;
                    float vA = Ab[(long)ck * Mdim + cm];
                    As[i * LDA + j] = (_Float16)(ok ? vA * ascale : 0.f);
                }
            }
        }

        // ---------------- stage B -> Bs[n][k] (f16) ----------------
        if (!BT) {
            if (n0 + BN <= Ndim && k0 + BK <= Kdim) {
                #pragma unroll
                for (int it = 0; it < 4; ++it) {         // 512 vec4 / 128 thr
                    int id = it * 128 + tid;
                    int j = id >> 4, n4 = id & 15;
                    v4f vv = *(const v4f*)(Bb + (long)(k0 + j) * Ndim + n0 + n4 * 4);
                    Bs[(n4 * 4 + 0) * LDB + j] = (_Float16)vv[0];
                    Bs[(n4 * 4 + 1) * LDB + j] = (_Float16)vv[1];
                    Bs[(n4 * 4 + 2) * LDB + j] = (_Float16)vv[2];
                    Bs[(n4 * 4 + 3) * LDB + j] = (_Float16)vv[3];
                }
            } else {
                #pragma unroll
                for (int it = 0; it < 16; ++it) {
                    int id = it * 128 + tid;
                    int j = id >> 6, n = id & 63;
                    int gk = k0 + j, gn = n0 + n;
                    bool ok = (gk < Kdim) && (gn < Ndim);
                    int ckk = gk < Kdim ? gk : Kdim - 1;
                    int cn  = gn < Ndim ? gn : Ndim - 1;
                    float vB = Bb[(long)ckk * Ndim + cn];
                    Bs[n * LDB + j] = (_Float16)(ok ? vB : 0.f);
                }
            }
        } else { // BT: B stored Ndim x Kdim; vectorize along k
            if (n0 + BN <= Ndim && k0 + BK <= Kdim) {
                #pragma unroll
                for (int it = 0; it < 4; ++it) {
                    int id = it * 128 + tid;
                    int n = id >> 3, j4 = id & 7;
                    v4f vv = *(const v4f*)(Bb + (long)(n0 + n) * Kdim + k0 + j4 * 4);
                    v4h hh;
                    hh[0] = (_Float16)vv[0]; hh[1] = (_Float16)vv[1];
                    hh[2] = (_Float16)vv[2]; hh[3] = (_Float16)vv[3];
                    *(v4h*)&Bs[n * LDB + j4 * 4] = hh;
                }
            } else {
                #pragma unroll
                for (int it = 0; it < 16; ++it) {
                    int id = it * 128 + tid;
                    int n = id >> 5, j = id & 31;
                    int gk = k0 + j, gn = n0 + n;
                    bool ok = (gk < Kdim) && (gn < Ndim);
                    int ckk = gk < Kdim ? gk : Kdim - 1;
                    int cn  = gn < Ndim ? gn : Ndim - 1;
                    float vB = Bb[(long)cn * Kdim + ckk];
                    Bs[n * LDB + j] = (_Float16)(ok ? vB : 0.f);
                }
            }
        }
        __syncthreads();   // staging complete; rawA free for next async tile

        if (pipeA && (k0 + BK < Kdim)) {
            #pragma unroll
            for (int it = 0; it < 8; ++it) {
                int id = it * 128 + tid;
                int i = id >> 3, j4 = id & 7;
                async_b128(&rawA[id * 4], Ab + (long)(m0 + i) * Kdim + (k0 + BK) + j4 * 4);
            }
        }

        // ---------------- WMMA: wave -> rows [wave*32, +32) x 64 cols ----------
        {
            const int kb = (lane < 16) ? 0 : 8;
            v16h afr[2];
            #pragma unroll
            for (int rt = 0; rt < 2; ++rt) {
                const _Float16* pa = &As[(wave * 32 + rt * 16 + (lane & 15)) * LDA];
                v8h alo = *(const v8h*)(pa + kb);
                v8h ahi = *(const v8h*)(pa + 16 + kb);
                #pragma unroll
                for (int i = 0; i < 8; ++i) { afr[rt][i] = alo[i]; afr[rt][8 + i] = ahi[i]; }
            }
            #pragma unroll
            for (int nt = 0; nt < 4; ++nt) {
                const _Float16* pb = &Bs[(nt * 16 + (lane & 15)) * LDB + ((lane < 16) ? 0 : 8)];
                v8h blo = *(const v8h*)pb;
                v8h bhi = *(const v8h*)(pb + 16);
                v16h bf;
                #pragma unroll
                for (int i = 0; i < 8; ++i) { bf[i] = blo[i]; bf[8 + i] = bhi[i]; }
                acc[0][nt] = __builtin_amdgcn_wmma_f32_16x16x32_f16(
                    false, afr[0], false, bf, (short)0, acc[0][nt], false, false);
                acc[1][nt] = __builtin_amdgcn_wmma_f32_16x16x32_f16(
                    false, afr[1], false, bf, (short)0, acc[1][nt], false, false);
            }
        }
        __syncthreads();
    }

    // ---- epilogue: C/D layout: lane l -> col (l&15), rows r + (l<16?0:8) ----
    const int lr   = lane & 15;
    const int radd = (lane < 16) ? 0 : 8;
    #pragma unroll
    for (int rt = 0; rt < 2; ++rt) {
        #pragma unroll
        for (int nt = 0; nt < 4; ++nt) {
            int gn = n0 + nt * 16 + lr;
            if (gn >= Ndim) continue;
            float bv = bias ? bias[gn] : 0.f;
            #pragma unroll
            for (int r = 0; r < 8; ++r) {
                int gm = m0 + wave * 32 + rt * 16 + radd + r;
                if (gm >= Mdim) continue;
                float vout = acc[rt][nt][r] + bv;
                if (rowscale) vout *= rowscale[(long)z * Mdim + gm];
                long oidx;
                if (OUTMODE == 0) {
                    oidx = (long)z * sC + (long)gm * Ndim + gn;
                } else if (OUTMODE == 1) {
                    int bidx = gm / Nseq, nn = gm % Nseq;
                    int hh = gn >> 6, dd = gn & 63;
                    oidx = (((long)(bidx * Hm + hh)) * Nseq + nn) * 64 + dd;
                } else {
                    oidx = (long)gm * ((long)Hm * Ndim) + (long)z * Ndim + gn;
                }
                if (resid) vout += resid[oidx];
                C[oidx] = vout;
            }
        }
    }
}

// ---------------------------------------------------------------------------
// Conv1d(k=3, pad=1) + eval BatchNorm + ReLU
// ---------------------------------------------------------------------------
__global__ __launch_bounds__(256) void conv1_kernel(
    const float* __restrict__ x, const float* __restrict__ wgt,
    const float* __restrict__ bias, const float* __restrict__ g,
    const float* __restrict__ beta, const float* __restrict__ mean,
    const float* __restrict__ var, float* __restrict__ out)
{
    long idx = (long)blockIdx.x * 256 + threadIdx.x;   // (b, co, t), t fastest
    if (idx >= (long)N_BATCH * 128 * SEQ) return;
    int t  = (int)(idx % SEQ);
    long r = idx / SEQ;
    int co = (int)(r % 128);
    int b  = (int)(r / 128);
    float acc = 0.f;
    #pragma unroll
    for (int kt = 0; kt < 3; ++kt) {
        int tt = t + kt - 1;
        if (tt < 0 || tt >= SEQ) continue;
        const float* xp = x + ((long)b * SEQ + tt) * N_MELS;   // x is (B,T,NMELS)
        const float* wp = wgt + (long)co * N_MELS * 3 + kt;
        for (int ci = 0; ci < N_MELS; ++ci) acc += xp[ci] * wp[ci * 3];
    }
    acc += bias[co];
    acc = (acc - mean[co]) * rsqrtf(var[co] + 1e-5f) * g[co] + beta[co];
    out[idx] = fmaxf(acc, 0.f);
}

__global__ __launch_bounds__(256) void conv2_kernel(
    const float* __restrict__ c1, const float* __restrict__ wgt,
    const float* __restrict__ bias, const float* __restrict__ g,
    const float* __restrict__ beta, const float* __restrict__ mean,
    const float* __restrict__ var, float* __restrict__ outT)
{
    long idx = (long)blockIdx.x * 256 + threadIdx.x;   // (b, t, co), co fastest
    if (idx >= (long)N_BATCH * SEQ * DIM) return;
    int co = (int)(idx % DIM);
    long r = idx / DIM;
    int t = (int)(r % SEQ);
    int b = (int)(r / SEQ);
    float acc = 0.f;
    for (int kt = 0; kt < 3; ++kt) {
        int tt = t + kt - 1;
        if (tt < 0 || tt >= SEQ) continue;
        const float* cp = c1 + ((long)b * 128) * SEQ + tt;     // (B,128,T)
        const float* wp = wgt + (long)co * 128 * 3 + kt;
        for (int ci = 0; ci < 128; ++ci) acc += cp[(long)ci * SEQ] * wp[ci * 3];
    }
    acc += bias[co];
    acc = (acc - mean[co]) * rsqrtf(var[co] + 1e-5f) * g[co] + beta[co];
    outT[idx] = fmaxf(acc, 0.f);   // stored transposed: (B, T, 256)
}

// ---------------------------------------------------------------------------
// LayerNorm over last dim RL (optional ReLU). One block(128) per row.
// ---------------------------------------------------------------------------
__global__ __launch_bounds__(128) void ln_kernel(
    const float* __restrict__ xin, const float* __restrict__ g,
    const float* __restrict__ b, float* __restrict__ y, int RL, int relu)
{
    __shared__ float sh[128];
    long row = blockIdx.x;
    const float* xr = xin + row * RL;
    float* yr = y + row * RL;
    float s = 0.f;
    for (int j = threadIdx.x; j < RL; j += 128) s += xr[j];
    sh[threadIdx.x] = s; __syncthreads();
    for (int st = 64; st > 0; st >>= 1) { if (threadIdx.x < st) sh[threadIdx.x] += sh[threadIdx.x + st]; __syncthreads(); }
    float mu = sh[0] / RL; __syncthreads();
    float vs = 0.f;
    for (int j = threadIdx.x; j < RL; j += 128) { float d = xr[j] - mu; vs += d * d; }
    sh[threadIdx.x] = vs; __syncthreads();
    for (int st = 64; st > 0; st >>= 1) { if (threadIdx.x < st) sh[threadIdx.x] += sh[threadIdx.x + st]; __syncthreads(); }
    float rstd = rsqrtf(sh[0] / RL + 1e-5f);
    for (int j = threadIdx.x; j < RL; j += 128) {
        float o = (xr[j] - mu) * rstd * g[j] + b[j];
        if (relu) o = fmaxf(o, 0.f);
        yr[j] = o;
    }
}

__global__ __launch_bounds__(256) void posadd_copy(
    const float* __restrict__ pos, float* __restrict__ x1, float* __restrict__ x2)
{
    long i = (long)blockIdx.x * 256 + threadIdx.x;
    if (i >= (long)ROWS_TOT * DIM) return;
    long nrow = (i >> 8) % SEQ;
    int c = (int)(i & 255);
    float v = x1[i] + pos[nrow * DIM + c];
    x1[i] = v; x2[i] = v;
}

__global__ __launch_bounds__(256) void diag_kernel(
    const float* __restrict__ data, float* __restrict__ dg, int rows)
{
    int r = blockIdx.x * 256 + threadIdx.x;
    if (r >= rows) return;
    const float* p = data + (long)r * DHEAD;
    float s = 0.f;
    #pragma unroll 8
    for (int j = 0; j < DHEAD; ++j) s += p[j] * p[j];
    dg[r] = 0.0625f * s;   // 0.5 * dn^2 = 0.5 * 64^-0.5
}

__global__ __launch_bounds__(128) void qp_transform(
    float* __restrict__ dd, const float* __restrict__ dg)
{
    __shared__ float sh[128];
    long row = blockIdx.x;
    float* p = dd + row * MFEAT;
    float mx = -3.4e38f;
    for (int j = threadIdx.x; j < MFEAT; j += 128) mx = fmaxf(mx, p[j]);
    sh[threadIdx.x] = mx; __syncthreads();
    for (int st = 64; st > 0; st >>= 1) { if (threadIdx.x < st) sh[threadIdx.x] = fmaxf(sh[threadIdx.x], sh[threadIdx.x + st]); __syncthreads(); }
    mx = sh[0];
    float d = dg[row];
    for (int j = threadIdx.x; j < MFEAT; j += 128)
        p[j] = RATIO * (expf(p[j] - d - mx) + EPS_KF);
}

__global__ __launch_bounds__(256) void kp_transform(
    float* __restrict__ dd, const float* __restrict__ dg, const float* __restrict__ gmax)
{
    long i = (long)blockIdx.x * 256 + threadIdx.x;
    if (i >= (long)HEADS * SEQ * MFEAT) return;
    long row = i >> 9;
    dd[i] = RATIO * (expf(dd[i] - dg[row] - gmax[0]) + EPS_KF);
}

__global__ __launch_bounds__(256) void max_partial(
    const float* __restrict__ xin, long n, float* __restrict__ outp)
{
    __shared__ float sh[256];
    float mx = -3.4e38f;
    for (long i = (long)blockIdx.x * 256 + threadIdx.x; i < n; i += (long)gridDim.x * 256)
        mx = fmaxf(mx, xin[i]);
    sh[threadIdx.x] = mx; __syncthreads();
    for (int st = 128; st > 0; st >>= 1) { if (threadIdx.x < st) sh[threadIdx.x] = fmaxf(sh[threadIdx.x], sh[threadIdx.x + st]); __syncthreads(); }
    if (threadIdx.x == 0) outp[blockIdx.x] = sh[0];
}

__global__ __launch_bounds__(256) void max_final(
    const float* __restrict__ xin, int n, float* __restrict__ outp)
{
    __shared__ float sh[256];
    float mx = -3.4e38f;
    for (int i = threadIdx.x; i < n; i += 256) mx = fmaxf(mx, xin[i]);
    sh[threadIdx.x] = mx; __syncthreads();
    for (int st = 128; st > 0; st >>= 1) { if (threadIdx.x < st) sh[threadIdx.x] = fmaxf(sh[threadIdx.x], sh[threadIdx.x + st]); __syncthreads(); }
    if (threadIdx.x == 0) outp[0] = sh[0];
}

__global__ __launch_bounds__(256) void ksum_kernel(
    const float* __restrict__ kp, float* __restrict__ ks)
{
    int h = blockIdx.x >> 1;
    int m = ((blockIdx.x & 1) << 8) + threadIdx.x;
    const float* p = kp + (long)h * SEQ * MFEAT + m;
    float s = 0.f;
    for (int n = 0; n < SEQ; ++n) s += p[(long)n * MFEAT];
    ks[h * MFEAT + m] = s;
}

__global__ __launch_bounds__(128) void dinv_kernel(
    const float* __restrict__ qp, const float* __restrict__ ks, float* __restrict__ dv)
{
    __shared__ float sh[128];
    long row = blockIdx.x;               // h*SEQ + n
    int h = (int)(row / SEQ);
    const float* p = qp + row * MFEAT;
    const float* kk = ks + (long)h * MFEAT;
    float s = 0.f;
    for (int j = threadIdx.x; j < MFEAT; j += 128) s += p[j] * kk[j];
    sh[threadIdx.x] = s; __syncthreads();
    for (int st = 64; st > 0; st >>= 1) { if (threadIdx.x < st) sh[threadIdx.x] += sh[threadIdx.x + st]; __syncthreads(); }
    if (threadIdx.x == 0) dv[row] = 1.f / sh[0];
}

__global__ __launch_bounds__(256) void glu_kernel(
    const float* __restrict__ hb, float* __restrict__ outg, long rows)
{
    long i = (long)blockIdx.x * 256 + threadIdx.x;
    if (i >= rows * FF_HID) return;
    long r = i >> 10;
    int j = (int)(i & 1023);
    float u  = hb[r * 2048 + j];
    float vv = hb[r * 2048 + 1024 + j];
    float ge = 0.5f * u * (1.f + erff(u * 0.7071067811865475f));  // exact GELU
    outg[i] = ge * vv;
}

__global__ __launch_bounds__(256) void add_kernel(
    const float* __restrict__ a, const float* __restrict__ b2,
    float* __restrict__ o, long n)
{
    long i = (long)blockIdx.x * 256 + threadIdx.x;
    if (i < n) o[i] = a[i] + b2[i];
}

__global__ __launch_bounds__(256) void clf2_kernel(
    const float* __restrict__ h, const float* __restrict__ w,
    const float* __restrict__ b, float* __restrict__ out)
{
    int wid = threadIdx.x >> 5, lane = threadIdx.x & 31;
    long row = (long)blockIdx.x * 8 + wid;
    if (row >= ROWS_TOT) return;
    const float* p = h + row * 128;
    float s = 0.f;
    for (int j = lane; j < 128; j += 32) s += p[j] * w[j];
    for (int off = 16; off > 0; off >>= 1) s += __shfl_down(s, off, 32);
    if (lane == 0) out[row] = s + b[0];
}

// ---------------------------------------------------------------------------
extern "C" void kernel_launch(void* const* d_in, const int* in_sizes, int n_in,
                              void* d_out, int out_size, void* d_ws, size_t ws_size,
                              hipStream_t stream)
{
    (void)in_sizes; (void)n_in; (void)out_size; (void)ws_size;
    const float* xin   = (const float*)d_in[0];
    const float* c1w   = (const float*)d_in[1];
    const float* c1b   = (const float*)d_in[2];
    const float* bn1g  = (const float*)d_in[3];
    const float* bn1b  = (const float*)d_in[4];
    const float* bn1m  = (const float*)d_in[5];
    const float* bn1v  = (const float*)d_in[6];
    const float* c2w   = (const float*)d_in[7];
    const float* c2b   = (const float*)d_in[8];
    const float* bn2g  = (const float*)d_in[9];
    const float* bn2b  = (const float*)d_in[10];
    const float* bn2m  = (const float*)d_in[11];
    const float* bn2v  = (const float*)d_in[12];
    const float* projw = (const float*)d_in[13];
    const float* projb = (const float*)d_in[14];
    const float* ln0g  = (const float*)d_in[15];
    const float* ln0b  = (const float*)d_in[16];
    const float* pos   = (const float*)d_in[17];
    const float* pmat  = (const float*)d_in[18];
    const float* clfw1 = (const float*)d_in[83];
    const float* clfb1 = (const float*)d_in[84];
    const float* clfg  = (const float*)d_in[85];
    const float* clfbt = (const float*)d_in[86];
    const float* clfw2 = (const float*)d_in[87];
    const float* clfb2 = (const float*)d_in[88];
    float* out = (float*)d_out;
    auto P = [&](int l, int j) { return (const float*)d_in[19 + l * 16 + j]; };

    // -------- workspace arena (floats) --------
    float* w = (float*)d_ws;
    const long SZ = (long)ROWS_TOT * DIM;          // 6,144,000 floats
    float* x1    = w;  w += SZ;
    float* x2    = w;  w += SZ;
    float* xln   = w;  w += SZ;
    float* qb    = w;  w += SZ;                    // (B,H,N,DH)
    float* kb    = w;  w += SZ;
    float* vb    = w;  w += SZ;
    float* t0    = w;  w += SZ;                    // attn-out / proj tmp / clf tmp
    float* qp    = w;  w += SZ;                    // per-batch (H,N,M)
    float* kp    = w;  w += SZ;
    float* ffh   = w;  w += 2 * SZ;                // 6000 x 2048 chunk
    float* glu   = w;  w += SZ;                    // 6000 x 1024 chunk
    float* c1o   = w;  w += (long)N_BATCH * 128 * SEQ;
    float* ctxb  = w;  w += (long)HEADS * MFEAT * DHEAD;
    float* ksum  = w;  w += HEADS * MFEAT;
    float* dinv  = w;  w += (long)N_BATCH * HEADS * SEQ;
    float* diagq = w;  w += HEADS * SEQ;
    float* diagk = w;  w += HEADS * SEQ;
    float* kred  = w;  w += 1024 + 16;             // partials + per-b maxima + gmax

    const long BHS = (long)HEADS * SEQ * DHEAD;    // per-batch q/k/v stride
    const int MBr = cdiv(ROWS_TOT, 128);           // 188
    const int MBs = cdiv(SEQ, 128);                // 24
    const int MBf = cdiv(6000, 128);               // 47

    // -------- conv front-end --------
    conv1_kernel<<<cdiv((long)N_BATCH * 128 * SEQ, 256), 256, 0, stream>>>(
        xin, c1w, c1b, bn1g, bn1b, bn1m, bn1v, c1o);
    conv2_kernel<<<cdiv((long)N_BATCH * SEQ * DIM, 256), 256, 0, stream>>>(
        c1o, c2w, c2b, bn2g, bn2b, bn2m, bn2v, xln);
    gemm_wmma<false, false, 0><<<dim3(DIM / 64, MBr, 1), 128, 0, stream>>>(
        xln, projw, projb, nullptr, nullptr, t0, ROWS_TOT, DIM, DIM, 0, 0, 0, 0, 0, 1.f);
    ln_kernel<<<ROWS_TOT, 128, 0, stream>>>(t0, ln0g, ln0b, x1, DIM, 0);
    posadd_copy<<<cdiv(SZ, 256), 256, 0, stream>>>(pos, x1, x2);

    // -------- reversible layers --------
    for (int l = 0; l < N_LAYERS; ++l) {
        // attention branch: y1 = x1 + attn(LN(x2))
        ln_kernel<<<ROWS_TOT, 128, 0, stream>>>(x2, P(l, 0), P(l, 1), xln, DIM, 0);
        gemm_wmma<false, false, 1><<<dim3(4, MBr, 1), 128, 0, stream>>>(
            xln, P(l, 2), P(l, 3), nullptr, nullptr, qb, ROWS_TOT, DIM, DIM, 0, 0, 0, HEADS, SEQ, 1.f);
        gemm_wmma<false, false, 1><<<dim3(4, MBr, 1), 128, 0, stream>>>(
            xln, P(l, 4), P(l, 5), nullptr, nullptr, kb, ROWS_TOT, DIM, DIM, 0, 0, 0, HEADS, SEQ, 1.f);
        gemm_wmma<false, false, 1><<<dim3(4, MBr, 1), 128, 0, stream>>>(
            xln, P(l, 6), P(l, 7), nullptr, nullptr, vb, ROWS_TOT, DIM, DIM, 0, 0, 0, HEADS, SEQ, 1.f);

        // pass 1: global key-feature max over (B,H,N,M)
        for (int b = 0; b < N_BATCH; ++b) {
            gemm_wmma<false, true, 0><<<dim3(MFEAT / 64, MBs, HEADS), 128, 0, stream>>>(
                kb + (long)b * BHS, pmat, nullptr, nullptr, nullptr, kp,
                SEQ, MFEAT, DHEAD, (long)SEQ * DHEAD, 0, (long)SEQ * MFEAT, 0, 0, DN_SCALE);
            max_partial<<<1024, 256, 0, stream>>>(kp, (long)HEADS * SEQ * MFEAT, kred);
            max_final<<<1, 256, 0, stream>>>(kred, 1024, kred + 1024 + b);
        }
        max_final<<<1, 256, 0, stream>>>(kred + 1024, N_BATCH, kred + 1033);

        // pass 2: per-batch FAVOR+ attention
        for (int b = 0; b < N_BATCH; ++b) {
            const float* qbb = qb + (long)b * BHS;
            const float* kbb = kb + (long)b * BHS;
            const float* vbb = vb + (long)b * BHS;
            gemm_wmma<false, true, 0><<<dim3(MFEAT / 64, MBs, HEADS), 128, 0, stream>>>(
                qbb, pmat, nullptr, nullptr, nullptr, qp,
                SEQ, MFEAT, DHEAD, (long)SEQ * DHEAD, 0, (long)SEQ * MFEAT, 0, 0, DN_SCALE);
            gemm_wmma<false, true, 0><<<dim3(MFEAT / 64, MBs, HEADS), 128, 0, stream>>>(
                kbb, pmat, nullptr, nullptr, nullptr, kp,
                SEQ, MFEAT, DHEAD, (long)SEQ * DHEAD, 0, (long)SEQ * MFEAT, 0, 0, DN_SCALE);
            diag_kernel<<<cdiv(HEADS * SEQ, 256), 256, 0, stream>>>(qbb, diagq, HEADS * SEQ);
            diag_kernel<<<cdiv(HEADS * SEQ, 256), 256, 0, stream>>>(kbb, diagk, HEADS * SEQ);
            qp_transform<<<HEADS * SEQ, 128, 0, stream>>>(qp, diagq);
            kp_transform<<<cdiv((long)HEADS * SEQ * MFEAT, 256), 256, 0, stream>>>(kp, diagk, kred + 1033);
            ksum_kernel<<<HEADS * 2, 256, 0, stream>>>(kp, ksum);
            // ctx[h] = kp[h]^T (MxN) @ v[h] (NxDH)
            gemm_wmma<true, false, 0><<<dim3(1, MFEAT / 128, HEADS), 128, 0, stream>>>(
                kp, vbb, nullptr, nullptr, nullptr, ctxb, MFEAT, DHEAD, SEQ,
                (long)SEQ * MFEAT, (long)SEQ * DHEAD, (long)MFEAT * DHEAD, 0, 0, 1.f);
            dinv_kernel<<<HEADS * SEQ, 128, 0, stream>>>(qp, ksum, dinv + (long)b * HEADS * SEQ);
            // out[h] = (qp[h] @ ctx[h]) * dinv, stored merged-head (N, DIM)
            gemm_wmma<false, false, 2><<<dim3(1, MBs, HEADS), 128, 0, stream>>>(
                qp, ctxb, nullptr, nullptr, dinv + (long)b * HEADS * SEQ,
                t0 + (long)b * SEQ * DIM, SEQ, DHEAD, MFEAT,
                (long)SEQ * MFEAT, (long)MFEAT * DHEAD, 0, HEADS, SEQ, 1.f);
        }
        // O projection + residual: x1 <- x1 + attnout @ wo + bo
        gemm_wmma<false, false, 0><<<dim3(4, MBr, 1), 128, 0, stream>>>(
            t0, P(l, 8), P(l, 9), x1, nullptr, x1, ROWS_TOT, DIM, DIM, 0, 0, 0, 0, 0, 1.f);

        // FFN branch: x2 <- x2 + GLU_FF(LN(x1)), processed in 4 row chunks
        ln_kernel<<<ROWS_TOT, 128, 0, stream>>>(x1, P(l, 10), P(l, 11), xln, DIM, 0);
        for (int c = 0; c < 4; ++c) {
            long r0 = (long)c * 6000;
            gemm_wmma<false, false, 0><<<dim3(2 * FF_HID / 64, MBf, 1), 128, 0, stream>>>(
                xln + r0 * DIM, P(l, 12), P(l, 13), nullptr, nullptr, ffh,
                6000, 2 * FF_HID, DIM, 0, 0, 0, 0, 0, 1.f);
            glu_kernel<<<cdiv(6000L * FF_HID, 256), 256, 0, stream>>>(ffh, glu, 6000);
            gemm_wmma<false, false, 0><<<dim3(DIM / 64, MBf, 1), 128, 0, stream>>>(
                glu, P(l, 14), P(l, 15), x2 + r0 * DIM, nullptr, x2 + r0 * DIM,
                6000, DIM, FF_HID, 0, 0, 0, 0, 0, 1.f);
        }
    }

    // -------- classifier head --------
    add_kernel<<<cdiv(SZ, 256), 256, 0, stream>>>(x1, x2, xln, SZ);
    gemm_wmma<false, false, 0><<<dim3(128 / 64, MBr, 1), 128, 0, stream>>>(
        xln, clfw1, clfb1, nullptr, nullptr, t0, ROWS_TOT, 128, DIM, 0, 0, 0, 0, 0, 1.f);
    ln_kernel<<<ROWS_TOT, 128, 0, stream>>>(t0, clfg, clfbt, t0, 128, 1);
    clf2_kernel<<<cdiv(ROWS_TOT, 8), 256, 0, stream>>>(t0, clfw2, clfb2, out);
}
